// EnhancedAttentionGNNAutoencoder_8890582302925
// MI455X (gfx1250) — compile-verified, matrix operand order checked by reading, not run on hardware
//
#include <hip/hip_runtime.h>
#include <hip/hip_bf16.h>

typedef __attribute__((ext_vector_type(16))) _Float16 v16h;
typedef __attribute__((ext_vector_type(8)))  float    v8f;

// ---------------------------------------------------------------------------
// WMMA GEMM:  C[M,Nc] = A[M,K] * B[K,Nc]   (f32 in, f16 WMMA, f32 accumulate)
// Each wave32 computes a 16x64 output strip = 4 WMMA tiles, reusing one A
// fragment across 4 B fragments per K-step (4 back-to-back v_wmma per iter).
// M multiple of 16 (N=50000=3125*16), K multiple of 32 (64/128),
// Nc multiple of 64 (64/128/512).
// Fragment layout per CDNA5 ISA 7.12.2 (16-bit A 16x32): lane group g=lane/16,
// halves t=0..7 -> K = 8g+t ; t=8..15 -> K = 16+8g+(t-8). B mirrors A over N.
// A's two 8-long K-runs are loaded as 4x float4 (global_load_b128).
// ---------------------------------------------------------------------------
__global__ __launch_bounds__(256) void gat_gemm_wmma(
    const float* __restrict__ A, const float* __restrict__ B,
    float* __restrict__ C, int M, int K, int Nc)
{
    const int wave = threadIdx.x >> 5;
    const int lane = threadIdx.x & 31;
    const int stripsN = Nc >> 6;            // 64-wide output strips
    const int tilesM  = M >> 4;
    long strip = (long)blockIdx.x * 8 + wave;
    if (strip >= (long)tilesM * stripsN) return;   // wave-uniform (EXEC all-1 for WMMA)
    const int tm = (int)(strip / stripsN);
    const int tn = (int)(strip % stripsN);
    const int m0 = tm << 4, n0 = tn << 6;
    const int grp = lane >> 4;              // 0 or 1
    const int lr  = lane & 15;
    const int rowA = m0 + lr;               // this lane's A row

    v8f acc[4] = {v8f{}, v8f{}, v8f{}, v8f{}};

    for (int k0 = 0; k0 < K; k0 += 32) {
        const int kb0 = k0 + 8 * grp;       // 8 consecutive K values
        const int kb1 = kb0 + 16;           // 8 more consecutive K values
        const float* Ar = A + (long)rowA * K;
        const float4 a0 = *(const float4*)(Ar + kb0);
        const float4 a1 = *(const float4*)(Ar + kb0 + 4);
        const float4 a2 = *(const float4*)(Ar + kb1);
        const float4 a3 = *(const float4*)(Ar + kb1 + 4);
        v16h a;
        a[0]=(_Float16)a0.x;  a[1]=(_Float16)a0.y;  a[2]=(_Float16)a0.z;  a[3]=(_Float16)a0.w;
        a[4]=(_Float16)a1.x;  a[5]=(_Float16)a1.y;  a[6]=(_Float16)a1.z;  a[7]=(_Float16)a1.w;
        a[8]=(_Float16)a2.x;  a[9]=(_Float16)a2.y;  a[10]=(_Float16)a2.z; a[11]=(_Float16)a2.w;
        a[12]=(_Float16)a3.x; a[13]=(_Float16)a3.y; a[14]=(_Float16)a3.z; a[15]=(_Float16)a3.w;

        #pragma unroll
        for (int j = 0; j < 4; ++j) {
            const int colB = n0 + 16 * j + lr;   // this lane's B column for sub-tile j
            v16h b;
            #pragma unroll
            for (int t = 0; t < 8; ++t) b[t]     = (_Float16)B[(long)(kb0 + t) * Nc + colB];
            #pragma unroll
            for (int t = 0; t < 8; ++t) b[8 + t] = (_Float16)B[(long)(kb1 + t) * Nc + colB];
            acc[j] = __builtin_amdgcn_wmma_f32_16x16x32_f16(
                false, a, false, b, (short)0, acc[j], false, false);
        }
    }

    #pragma unroll
    for (int j = 0; j < 4; ++j) {
        const int col = n0 + 16 * j + lr;
        #pragma unroll
        for (int r = 0; r < 8; ++r) {
            const int m = m0 + r + 8 * grp;
            C[(long)m * Nc + col] = acc[j][r];
        }
    }
}

// ---------------------------------------------------------------------------
// Small helpers
// ---------------------------------------------------------------------------
__device__ inline void atomicMaxF(float* addr, float v) {
    unsigned int* u = (unsigned int*)addr;
    unsigned int old = *u;
    while (__uint_as_float(old) < v) {
        unsigned int assumed = old;
        old = atomicCAS(u, assumed, __float_as_uint(v));
        if (old == assumed) break;
    }
}

__device__ inline void edge_sd(const int* __restrict__ ei, int E, int e, int& s, int& d) {
    if (e < E) { s = ei[e]; d = ei[E + e]; }
    else       { s = d = e - E; }            // self loops appended
}

__global__ void fill_f(float* __restrict__ p, float v, long n) {
    long i = (long)blockIdx.x * blockDim.x + threadIdx.x;
    if (i < n) p[i] = v;
}

// es[n,h] = <hW[n,h,:], a_s[h,:]> ; ed likewise
__global__ void attn_coef(const float* __restrict__ hW,
                          const float* __restrict__ a_s, const float* __restrict__ a_d,
                          float* __restrict__ es, float* __restrict__ ed,
                          int N, int H, int C)
{
    int i = blockIdx.x * blockDim.x + threadIdx.x;   // i = n*H + h
    if (i >= N * H) return;
    int h = i % H;
    const float* hp = hW + (long)i * C;
    const float* as = a_s + h * C;
    const float* ad = a_d + h * C;
    float s = 0.f, d = 0.f;
    for (int c = 0; c < C; ++c) { float v = hp[c]; s += v * as[c]; d += v * ad[c]; }
    es[i] = s; ed[i] = d;
}

// e = leaky_relu(es[src]+ed[dst]); segment max over dst
__global__ void edge_logits_max(const int* __restrict__ ei, int E, int N, int H,
                                const float* __restrict__ es, const float* __restrict__ ed,
                                float* __restrict__ ebuf, float* __restrict__ m)
{
    int e = blockIdx.x * blockDim.x + threadIdx.x;
    int ET = E + N;
    if (e >= ET) return;
    int s, d; edge_sd(ei, E, e, s, d);
    for (int h = 0; h < H; ++h) {
        float v = es[s * H + h] + ed[d * H + h];
        v = v > 0.f ? v : 0.2f * v;
        ebuf[(long)e * H + h] = v;
        atomicMaxF(&m[d * H + h], v);
    }
}

// p = exp(e - m[dst]); segment sum over dst
__global__ void edge_exp_sum(const int* __restrict__ ei, int E, int N, int H,
                             const float* __restrict__ m,
                             float* __restrict__ ebuf, float* __restrict__ den)
{
    int e = blockIdx.x * blockDim.x + threadIdx.x;
    int ET = E + N;
    if (e >= ET) return;
    int s, d; edge_sd(ei, E, e, s, d);
    (void)s;
    for (int h = 0; h < H; ++h) {
        float p = __expf(ebuf[(long)e * H + h] - m[d * H + h]);
        ebuf[(long)e * H + h] = p;
        atomicAdd(&den[d * H + h], p);
    }
}

// acc[dst, c] += (1/H) * alpha_{e,h} * hW[src, h, c]   (head mean folded in)
__global__ void edge_scatter(const int* __restrict__ ei, int E, int N, int H, int C,
                             const float* __restrict__ ebuf, const float* __restrict__ den,
                             const float* __restrict__ hW, float* __restrict__ acc)
{
    int i = blockIdx.x * blockDim.x + threadIdx.x;   // i = e*H + h
    int ET = E + N;
    if (i >= ET * H) return;
    int e = i / H, h = i % H;
    int s, d; edge_sd(ei, E, e, s, d);
    float alpha = ebuf[i] / (den[d * H + h] + 1e-16f) * (1.0f / (float)H);
    const float* sp = hW + ((long)s * H + h) * C;
    float* dp = acc + (long)d * C;
    for (int c = 0; c < C; ++c) atomicAdd(&dp[c], alpha * sp[c]);
}

__global__ void bias_act(const float* __restrict__ acc, const float* __restrict__ b,
                         float* __restrict__ out, int N, int C, int relu)
{
    long i = (long)blockIdx.x * blockDim.x + threadIdx.x;
    if (i >= (long)N * C) return;
    int c = (int)(i % C);
    float v = acc[i] + b[c];
    if (relu) v = fmaxf(v, 0.f);
    out[i] = v;
}

// gate[n] = sum_c relu(t[n,c]+g_b1[c]) * g_w2[c] + g_b2
__global__ void gate_reduce(const float* __restrict__ tbuf, const float* __restrict__ g_b1,
                            const float* __restrict__ g_w2, const float* __restrict__ g_b2,
                            float* __restrict__ gate, int N, int C)
{
    int n = blockIdx.x * blockDim.x + threadIdx.x;
    if (n >= N) return;
    const float* t = tbuf + (long)n * C;
    float sum = 0.f;
    for (int c = 0; c < C; ++c) {
        float v = fmaxf(t[c] + g_b1[c], 0.f);
        sum += v * g_w2[c];
    }
    gate[n] = sum + g_b2[0];
}

__global__ void pool_max(const float* __restrict__ gate, const int* __restrict__ batch,
                         float* __restrict__ gmax, int N)
{
    int n = blockIdx.x * blockDim.x + threadIdx.x;
    if (n >= N) return;
    atomicMaxF(&gmax[batch[n]], gate[n]);
}

__global__ void pool_expsum(float* __restrict__ gate, const int* __restrict__ batch,
                            const float* __restrict__ gmax, float* __restrict__ gden, int N)
{
    int n = blockIdx.x * blockDim.x + threadIdx.x;
    if (n >= N) return;
    int b = batch[n];
    float p = __expf(gate[n] - gmax[b]);
    gate[n] = p;
    atomicAdd(&gden[b], p);
}

__global__ void pool_weighted(const float* __restrict__ p, const float* __restrict__ gden,
                              const int* __restrict__ batch, const float* __restrict__ h,
                              float* __restrict__ pooled, int N, int C)
{
    int n = blockIdx.x * blockDim.x + threadIdx.x;
    if (n >= N) return;
    int b = batch[n];
    float att = p[n] / (gden[b] + 1e-16f);
    const float* hp = h + (long)n * C;
    float* pp = pooled + (long)b * C;
    for (int c = 0; c < C; ++c) atomicAdd(&pp[c], att * hp[c]);
}

__global__ void pool_bcast(const float* __restrict__ pooled, const int* __restrict__ batch,
                           float* __restrict__ out, int N, int C)
{
    long i = (long)blockIdx.x * blockDim.x + threadIdx.x;
    if (i >= (long)N * C) return;
    int n = (int)(i / C), c = (int)(i % C);
    out[i] = pooled[(long)batch[n] * C + c];
}

// ---------------------------------------------------------------------------
// Orchestration
// ---------------------------------------------------------------------------
static inline int cdiv(long a, int b) { return (int)((a + b - 1) / b); }

extern "C" void kernel_launch(void* const* d_in, const int* in_sizes, int n_in,
                              void* d_out, int out_size, void* d_ws, size_t ws_size,
                              hipStream_t stream)
{
    const int Din = 128, H = 8, C = 64, HC = 512, G = 16;
    const int N  = in_sizes[0] / Din;
    const int E  = in_sizes[1] / 2;
    const int ET = E + N;

    const float* x    = (const float*)d_in[0];
    const int*   ei   = (const int*)  d_in[1];
    const int*   batch= (const int*)  d_in[2];
    const float* W_e0 = (const float*)d_in[3];
    const float* as_e0= (const float*)d_in[4];
    const float* ad_e0= (const float*)d_in[5];
    const float* b_e0 = (const float*)d_in[6];
    const float* W_e1 = (const float*)d_in[7];
    const float* as_e1= (const float*)d_in[8];
    const float* ad_e1= (const float*)d_in[9];
    const float* b_e1 = (const float*)d_in[10];
    const float* W_d0 = (const float*)d_in[11];
    const float* as_d0= (const float*)d_in[12];
    const float* ad_d0= (const float*)d_in[13];
    const float* b_d0 = (const float*)d_in[14];
    const float* W_d1 = (const float*)d_in[15];
    const float* as_d1= (const float*)d_in[16];
    const float* ad_d1= (const float*)d_in[17];
    const float* b_d1 = (const float*)d_in[18];
    const float* g_w1 = (const float*)d_in[19];
    const float* g_b1 = (const float*)d_in[20];
    const float* g_w2 = (const float*)d_in[21];
    const float* g_b2 = (const float*)d_in[22];

    // workspace layout (floats)
    float* ws = (float*)d_ws;
    long o = 0;
    float* hW   = ws + o; o += (long)N * HC;   // N x 512
    float* es   = ws + o; o += (long)N * H;
    float* ed   = ws + o; o += (long)N * H;
    float* m    = ws + o; o += (long)N * H;
    float* den  = ws + o; o += (long)N * H;
    float* ebuf = ws + o; o += (long)ET * H;
    float* acc  = ws + o; o += (long)N * 128;
    float* h1   = ws + o; o += (long)N * C;    // also reused as decoder h4
    float* h2   = ws + o; o += (long)N * C;
    float* h3   = ws + o; o += (long)N * C;
    float* tbuf = ws + o; o += (long)N * C;
    float* gate = ws + o; o += (long)N;
    float* gmax = ws + o; o += G;
    float* gden = ws + o; o += G;
    float* pooled = ws + o; o += (long)G * C;
    (void)ws_size; (void)n_in; (void)out_size;

    const int B256 = 256;
    // grid: one wave per 16x64 strip, 8 waves per block
    #define GEMM(A_, B_, C_, M_, K_, Nc_) \
        gat_gemm_wmma<<<cdiv((long)((M_)>>4) * ((Nc_)>>6), 8), B256, 0, stream>>>(A_, B_, C_, M_, K_, Nc_)

    // ---- GAT layer macro: hW -> (es,ed) -> edge softmax -> scatter -> bias/act ----
    #define GAT_ATTN(asrc, adst, Hh, Cc, bias, outp, relu_)                                         \
        do {                                                                                        \
            attn_coef<<<cdiv((long)N*(Hh), B256), B256, 0, stream>>>(hW, asrc, adst, es, ed, N, Hh, Cc); \
            fill_f<<<cdiv((long)N*(Hh), B256), B256, 0, stream>>>(m, -1e30f, (long)N*(Hh));         \
            fill_f<<<cdiv((long)N*(Hh), B256), B256, 0, stream>>>(den, 0.f, (long)N*(Hh));          \
            fill_f<<<cdiv((long)N*(Cc), B256), B256, 0, stream>>>(acc, 0.f, (long)N*(Cc));          \
            edge_logits_max<<<cdiv((long)ET, B256), B256, 0, stream>>>(ei, E, N, Hh, es, ed, ebuf, m); \
            edge_exp_sum<<<cdiv((long)ET, B256), B256, 0, stream>>>(ei, E, N, Hh, m, ebuf, den);    \
            edge_scatter<<<cdiv((long)ET*(Hh), B256), B256, 0, stream>>>(ei, E, N, Hh, Cc, ebuf, den, hW, acc); \
            bias_act<<<cdiv((long)N*(Cc), B256), B256, 0, stream>>>(acc, bias, outp, N, Cc, relu_); \
        } while (0)

    // ===== encoder layer 0: x[N,128] -> h1[N,64] (relu) =====
    GEMM(x, W_e0, hW, N, Din, HC);
    GAT_ATTN(as_e0, ad_e0, H, C, b_e0, h1, 1);

    // ===== encoder layer 1: h1[N,64] -> h2[N,64] =====
    GEMM(h1, W_e1, hW, N, C, HC);
    GAT_ATTN(as_e1, ad_e1, H, C, b_e1, h2, 0);

    // ===== global attention pooling over 16 graphs =====
    GEMM(h2, g_w1, tbuf, N, C, C);
    gate_reduce<<<cdiv(N, B256), B256, 0, stream>>>(tbuf, g_b1, g_w2, g_b2, gate, N, C);
    fill_f<<<1, 64, 0, stream>>>(gmax, -1e30f, G);
    fill_f<<<1, 64, 0, stream>>>(gden, 0.f, G);
    fill_f<<<cdiv((long)G*C, B256), B256, 0, stream>>>(pooled, 0.f, (long)G*C);
    pool_max    <<<cdiv(N, B256), B256, 0, stream>>>(gate, batch, gmax, N);
    pool_expsum <<<cdiv(N, B256), B256, 0, stream>>>(gate, batch, gmax, gden, N);
    pool_weighted<<<cdiv(N, B256), B256, 0, stream>>>(gate, gden, batch, h2, pooled, N, C);
    pool_bcast  <<<cdiv((long)N*C, B256), B256, 0, stream>>>(pooled, batch, h3, N, C);

    // ===== decoder layer 0 (H=1, C=64): h3 -> h1 (relu) =====
    GEMM(h3, W_d0, hW, N, C, C);
    GAT_ATTN(as_d0, ad_d0, 1, C, b_d0, h1, 1);

    // ===== decoder layer 1 (H=1, C=128): h1 -> d_out =====
    GEMM(h1, W_d1, hW, N, C, Din);
    GAT_ATTN(as_d1, ad_d1, 1, Din, b_d1, (float*)d_out, 0);

    #undef GAT_ATTN
    #undef GEMM
}
